// SNN3_27736898798058
// MI455X (gfx1250) — matrix operand order repaired and checked by reading, not compiled, not run on hardware
//
#include <hip/hip_runtime.h>
#include <hip/hip_bf16.h>
#include <stdint.h>

// Problem sizes (fixed by reference)
#define B_     256
#define N_IN   1024
#define N_HID  2048
#define N_OUT  256
#define T_     128
#define M1     (B_ * T_)          // 32768 fused (batch,time) rows

// WMMA vector types (probe-confirmed signatures)
typedef __attribute__((ext_vector_type(16))) __bf16 v16bf;
typedef __attribute__((ext_vector_type(8)))  float  v8f;

union Frag {
    uint4 q[2];   // 2 x 16B LDS loads
    v16bf v;      // bf16 A/B operand
};

__device__ __forceinline__ uint16_t bf16_hi_bits(float f) {
    return (uint16_t)(__float_as_uint(f) >> 16);        // truncate (hi part of split)
}

// split two f32 into packed bf16 hi-pair and lo-pair (residual) words
__device__ __forceinline__ void split2(float f0, float f1, uint32_t& hi, uint32_t& lo) {
    uint32_t u0 = __float_as_uint(f0), u1 = __float_as_uint(f1);
    uint16_t h0 = (uint16_t)(u0 >> 16), h1 = (uint16_t)(u1 >> 16);
    float r0 = f0 - __uint_as_float(u0 & 0xFFFF0000u);
    float r1 = f1 - __uint_as_float(u1 & 0xFFFF0000u);
    hi = (uint32_t)h0 | ((uint32_t)h1 << 16);
    lo = (uint32_t)bf16_hi_bits(r0) | ((uint32_t)bf16_hi_bits(r1) << 16);
}

#define LDK 40    // padded LDS row stride in halves: 32 data + 8 pad (80B, 16B-aligned)

// ---------------------------------------------------------------------------
// Kernel 1: CUR1[m, h] = X[m, k] @ W1[h, k]^T + b1[h],  m = b*T + t
//   X[m,k] = x[b][k][t]  (x is [B, N_IN, T], t contiguous)
//   split-bf16: 3 wmmas (hi*hi + hi*lo + lo*hi), f32 accumulate
// Software-pipelined, double-buffered LDS (one barrier per K-step):
//   iter ks: [b128 global loads for ks+1 -> regs] [24 wmma on buf ks&1]
//            [convert+store -> buf (ks+1)&1] barrier
// block = 256 thr (8 waves, 2M x 4N), tile 128M x 128N, K-step 32
// grid = (M1/128, N_HID/128); M-block == one batch index since T==128
// ---------------------------------------------------------------------------
__global__ __launch_bounds__(256)
void snn_gemm1(const float* __restrict__ x, const float* __restrict__ W1,
               const float* __restrict__ b1, float* __restrict__ cur1)
{
    __shared__ __align__(16) uint16_t As_hi[2][128 * LDK];   // 4 x 2 x 10KB = 80KB total
    __shared__ __align__(16) uint16_t As_lo[2][128 * LDK];
    __shared__ __align__(16) uint16_t Ws_hi[2][128 * LDK];
    __shared__ __align__(16) uint16_t Ws_lo[2][128 * LDK];

    const int tid  = threadIdx.x;
    const int lane = tid & 31;
    const int wid  = tid >> 5;      // 0..7
    const int wm   = wid & 1;       // M half   (64 rows)
    const int wn   = wid >> 1;      // N quarter(32 cols)

    const int bM = blockIdx.x;          // == batch b (T_ == 128)
    const int m0 = bM * 128;
    const int n0 = blockIdx.y * 128;

    const int col  = lane & 15;
    const int sel  = lane >> 4;         // K-half selector per ISA layout
    const int lrow = lane & 15;

    // accumulators initialized with bias (bias depends only on column)
    v8f acc[4][2];
#pragma unroll
    for (int j = 0; j < 2; ++j) {
        float bias = b1[n0 + wn * 32 + j * 16 + col];
#pragma unroll
        for (int i = 0; i < 4; ++i)
#pragma unroll
            for (int r = 0; r < 8; ++r) acc[i][j][r] = bias;
    }

    const float* xb = x + (size_t)bM * N_IN * T_;   // x[b][k][t]

    float4 areg[4];      // A tile: 4 x b128 per thread (4 consecutive t, one k)
    float4 wreg[4];      // W tile: 4 x b128 per thread (4 consecutive k, one n)

    auto load_regs = [&](int K0) {
#pragma unroll
        for (int it = 0; it < 4; ++it) {
            int q  = it * 256 + tid;                 // 0..1023 float4s
            int k  = q >> 5;                         // 0..31
            int mq = (q & 31) * 4;                   // t, multiple of 4
            areg[it] = *(const float4*)(xb + (size_t)(K0 + k) * T_ + mq);
        }
#pragma unroll
        for (int it = 0; it < 4; ++it) {
            int q  = it * 256 + tid;                 // 0..1023 float4s
            int n  = q >> 3;                         // 0..127
            int kq = (q & 7) * 4;                    // 0..28
            wreg[it] = *(const float4*)(W1 + (size_t)(n0 + n) * N_IN + K0 + kq);
        }
    };

    auto store_lds = [&](int bb) {
#pragma unroll
        for (int it = 0; it < 4; ++it) {
            int q  = it * 256 + tid;
            int k  = q >> 5;
            int mq = (q & 31) * 4;
            float f[4] = {areg[it].x, areg[it].y, areg[it].z, areg[it].w};
#pragma unroll
            for (int c = 0; c < 4; ++c) {
                uint32_t u = __float_as_uint(f[c]);
                As_hi[bb][(mq + c) * LDK + k] = (uint16_t)(u >> 16);
                float r = f[c] - __uint_as_float(u & 0xFFFF0000u);
                As_lo[bb][(mq + c) * LDK + k] = bf16_hi_bits(r);
            }
        }
#pragma unroll
        for (int it = 0; it < 4; ++it) {
            int q  = it * 256 + tid;
            int n  = q >> 3;
            int kq = (q & 7) * 4;
            uint32_t hi01, lo01, hi23, lo23;
            split2(wreg[it].x, wreg[it].y, hi01, lo01);
            split2(wreg[it].z, wreg[it].w, hi23, lo23);
            *(uint32_t*)&Ws_hi[bb][n * LDK + kq]     = hi01;
            *(uint32_t*)&Ws_hi[bb][n * LDK + kq + 2] = hi23;
            *(uint32_t*)&Ws_lo[bb][n * LDK + kq]     = lo01;
            *(uint32_t*)&Ws_lo[bb][n * LDK + kq + 2] = lo23;
        }
    };

    auto compute = [&](int bb) {
        // per-lane fragments (ISA 7.12.2 16-bit A 16x32 / B 32x16 layouts)
        Frag ah[4], al[4];
#pragma unroll
        for (int i = 0; i < 4; ++i) {
            const uint16_t* ph = &As_hi[bb][(wm * 64 + i * 16 + lrow) * LDK];
            const uint16_t* pl = &As_lo[bb][(wm * 64 + i * 16 + lrow) * LDK];
            ah[i].q[0] = *(const uint4*)(ph + sel * 8);        // K {0..7} / {8..15}
            ah[i].q[1] = *(const uint4*)(ph + 16 + sel * 8);   // K {16..23} / {24..31}
            al[i].q[0] = *(const uint4*)(pl + sel * 8);
            al[i].q[1] = *(const uint4*)(pl + 16 + sel * 8);
        }
        Frag bh[2], bl[2];
#pragma unroll
        for (int j = 0; j < 2; ++j) {
            const uint16_t* ph = &Ws_hi[bb][(wn * 32 + j * 16 + lrow) * LDK] + sel * 8;
            const uint16_t* pl = &Ws_lo[bb][(wn * 32 + j * 16 + lrow) * LDK] + sel * 8;
            bh[j].q[0] = *(const uint4*)(ph);
            bh[j].q[1] = *(const uint4*)(ph + 16);
            bl[j].q[0] = *(const uint4*)(pl);
            bl[j].q[1] = *(const uint4*)(pl + 16);
        }
#pragma unroll
        for (int i = 0; i < 4; ++i)
#pragma unroll
            for (int j = 0; j < 2; ++j) {
                acc[i][j] = __builtin_amdgcn_wmma_f32_16x16x32_bf16(
                    false, ah[i].v, false, bh[j].v, (short)0, acc[i][j], false, false);
                acc[i][j] = __builtin_amdgcn_wmma_f32_16x16x32_bf16(
                    false, ah[i].v, false, bl[j].v, (short)0, acc[i][j], false, false);
                acc[i][j] = __builtin_amdgcn_wmma_f32_16x16x32_bf16(
                    false, al[i].v, false, bh[j].v, (short)0, acc[i][j], false, false);
            }
    };

    // prologue
    load_regs(0);
    store_lds(0);
    __syncthreads();

    const int NS = N_IN / 32;                // 32 K-steps
    for (int ks = 0; ks < NS; ++ks) {
        int cur = ks & 1;
        if (ks + 1 < NS) load_regs((ks + 1) * 32);   // global loads overlap wmma below
        compute(cur);
        if (ks + 1 < NS) store_lds(cur ^ 1);
        __syncthreads();
    }

    // ---- store (C layout: VGPR r -> M=r (lanes 0-15) / M=8+r (lanes 16-31), N=col)
    const int rbase = sel * 8;
#pragma unroll
    for (int i = 0; i < 4; ++i)
#pragma unroll
        for (int j = 0; j < 2; ++j) {
            int n = n0 + wn * 32 + j * 16 + col;
#pragma unroll
            for (int r = 0; r < 8; ++r) {
                int m = m0 + wm * 64 + i * 16 + rbase + r;
                cur1[(size_t)m * N_HID + n] = acc[i][j][r];
            }
        }
}

// ---------------------------------------------------------------------------
// Kernel 2: per-(b,h) membrane-2 scan over t; emits bf16 spikes {0,1}
// SPK2 layout: [m, h] = [(b*T+t), h]  (ready as GEMM2 A-matrix)
// ---------------------------------------------------------------------------
__global__ __launch_bounds__(256)
void snn_scan2(const float* __restrict__ cur1, uint16_t* __restrict__ spk2,
               const float* __restrict__ beta2p, const float* __restrict__ thr2p)
{
    int idx = blockIdx.x * 256 + threadIdx.x;       // b*N_HID + h
    int b = idx >> 11;                              // / N_HID
    int h = idx & (N_HID - 1);
    float beta = fminf(fmaxf(beta2p[0], 0.f), 1.f);
    float thr  = thr2p[0];
    const float* cp = cur1 + (size_t)b * T_ * N_HID + h;
    uint16_t*    sp = spk2 + (size_t)b * T_ * N_HID + h;
    float mem = 0.f;
    const uint16_t one_bf16 = 0x3F80;
    for (int t = 0; t < T_; ++t) {
        float cur   = cp[(size_t)t * N_HID];
        float reset = (mem - thr > 0.f) ? thr : 0.f; // detached reset * thr
        mem = beta * mem + cur - reset;
        sp[(size_t)t * N_HID] = (mem - thr > 0.f) ? one_bf16 : (uint16_t)0;
    }
}

// ---------------------------------------------------------------------------
// Kernel 3: CUR2[m, o] = SPK2[m, h] @ W2[o, h]^T + b2[o]
// Spikes exact in bf16 -> only W2 split (2 wmmas). K = 2048.
// A-tile staged with GLOBAL_LOAD_ASYNC_TO_LDS_B128 (ASYNCcnt path) into the
// next LDS buffer while wmmas run on the current one; W2 register-pipelined.
// grid = (M1/128, N_OUT/128) = (256, 2)
// ---------------------------------------------------------------------------
__global__ __launch_bounds__(256)
void snn_gemm2(const uint16_t* __restrict__ spk2, const float* __restrict__ W2,
               const float* __restrict__ b2, float* __restrict__ cur2)
{
    __shared__ __align__(16) uint16_t As[2][128 * LDK];      // 3 x 2 x 10KB = 60KB
    __shared__ __align__(16) uint16_t Ws_hi[2][128 * LDK];
    __shared__ __align__(16) uint16_t Ws_lo[2][128 * LDK];

    const int tid  = threadIdx.x;
    const int lane = tid & 31;
    const int wid  = tid >> 5;
    const int wm   = wid & 1;
    const int wn   = wid >> 1;

    const int m0 = blockIdx.x * 128;
    const int o0 = blockIdx.y * 128;

    const int col  = lane & 15;
    const int sel  = lane >> 4;
    const int lrow = lane & 15;

    v8f acc[4][2];
#pragma unroll
    for (int j = 0; j < 2; ++j) {
        float bias = b2[o0 + wn * 32 + j * 16 + col];
#pragma unroll
        for (int i = 0; i < 4; ++i)
#pragma unroll
            for (int r = 0; r < 8; ++r) acc[i][j][r] = bias;
    }

    float4 wreg[4];

    // async copy of spike tile: 512 b128 transfers, 2 per thread, VGPR-free
    auto async_a = [&](int bb, int K0) {
#pragma unroll
        for (int it = 0; it < 2; ++it) {
            int p  = it * 256 + tid;                 // 0..511
            int m  = p >> 2;
            int kp = (p & 3) * 8;                    // 8 halves = 16B per transfer
            const uint16_t* g = spk2 + (size_t)(m0 + m) * N_HID + K0 + kp;
            uint32_t loff = (uint32_t)(uintptr_t)&As[bb][m * LDK + kp];
            asm volatile("global_load_async_to_lds_b128 %0, %1, off"
                         :: "v"(loff), "v"((uint64_t)(uintptr_t)g)
                         : "memory");
        }
    };
    auto load_wregs = [&](int K0) {
#pragma unroll
        for (int it = 0; it < 4; ++it) {
            int q  = it * 256 + tid;                 // 0..1023 float4s
            int n  = q >> 3;
            int kq = (q & 7) * 4;
            wreg[it] = *(const float4*)(W2 + (size_t)(o0 + n) * N_HID + K0 + kq);
        }
    };
    auto store_w = [&](int bb) {
#pragma unroll
        for (int it = 0; it < 4; ++it) {
            int q  = it * 256 + tid;
            int n  = q >> 3;
            int kq = (q & 7) * 4;
            uint32_t hi01, lo01, hi23, lo23;
            split2(wreg[it].x, wreg[it].y, hi01, lo01);
            split2(wreg[it].z, wreg[it].w, hi23, lo23);
            *(uint32_t*)&Ws_hi[bb][n * LDK + kq]     = hi01;
            *(uint32_t*)&Ws_hi[bb][n * LDK + kq + 2] = hi23;
            *(uint32_t*)&Ws_lo[bb][n * LDK + kq]     = lo01;
            *(uint32_t*)&Ws_lo[bb][n * LDK + kq + 2] = lo23;
        }
    };
    auto compute = [&](int bb) {
        Frag a[4];
#pragma unroll
        for (int i = 0; i < 4; ++i) {
            const uint16_t* pb = &As[bb][(wm * 64 + i * 16 + lrow) * LDK];
            a[i].q[0] = *(const uint4*)(pb + sel * 8);
            a[i].q[1] = *(const uint4*)(pb + 16 + sel * 8);
        }
        Frag bh[2], bl[2];
#pragma unroll
        for (int j = 0; j < 2; ++j) {
            const uint16_t* ph = &Ws_hi[bb][(wn * 32 + j * 16 + lrow) * LDK] + sel * 8;
            const uint16_t* pl = &Ws_lo[bb][(wn * 32 + j * 16 + lrow) * LDK] + sel * 8;
            bh[j].q[0] = *(const uint4*)(ph);
            bh[j].q[1] = *(const uint4*)(ph + 16);
            bl[j].q[0] = *(const uint4*)(pl);
            bl[j].q[1] = *(const uint4*)(pl + 16);
        }
#pragma unroll
        for (int i = 0; i < 4; ++i)
#pragma unroll
            for (int j = 0; j < 2; ++j) {
                acc[i][j] = __builtin_amdgcn_wmma_f32_16x16x32_bf16(
                    false, a[i].v, false, bh[j].v, (short)0, acc[i][j], false, false);
                acc[i][j] = __builtin_amdgcn_wmma_f32_16x16x32_bf16(
                    false, a[i].v, false, bl[j].v, (short)0, acc[i][j], false, false);
            }
    };

    // prologue: fill buffer 0
    async_a(0, 0);
    load_wregs(0);
    store_w(0);
    asm volatile("s_wait_asynccnt 0x0" ::: "memory");
    __syncthreads();

    const int NS = N_HID / 32;               // 64 K-steps
    for (int ks = 0; ks < NS; ++ks) {
        int cur = ks & 1;
        if (ks + 1 < NS) {
            async_a(cur ^ 1, (ks + 1) * 32);     // async spikes -> next buffer
            load_wregs((ks + 1) * 32);           // W2 f32 -> regs
        }
        compute(cur);                            // 16 wmmas on current buffer
        if (ks + 1 < NS) store_w(cur ^ 1);       // convert W2 -> next buffer
        asm volatile("s_wait_asynccnt 0x0" ::: "memory");
        __syncthreads();
    }

    const int rbase = sel * 8;
#pragma unroll
    for (int i = 0; i < 4; ++i)
#pragma unroll
        for (int j = 0; j < 2; ++j) {
            int n = o0 + wn * 32 + j * 16 + col;
#pragma unroll
            for (int r = 0; r < 8; ++r) {
                int m = m0 + wm * 64 + i * 16 + rbase + r;
                cur2[(size_t)m * N_OUT + n] = acc[i][j][r];
            }
        }
}

// ---------------------------------------------------------------------------
// Kernel 4: per-(b,o) membrane-3 scan; records mem3 to out[B, N_OUT, T]
// LDS-staged so the [.., t]-contiguous output is written coalesced.
// grid = (B_), block = 256 (one thread per o)
// ---------------------------------------------------------------------------
__global__ __launch_bounds__(256)
void snn_scan3(const float* __restrict__ cur2, float* __restrict__ out,
               const float* __restrict__ beta3p, const float* __restrict__ thr3p)
{
    __shared__ float buf[256 * 32];       // [o][t-chunk of 32] = 32KB
    const int b = blockIdx.x;
    const int o = threadIdx.x;
    float beta = fminf(fmaxf(beta3p[0], 0.f), 1.f);
    float thr  = thr3p[0];
    const float* cp = cur2 + (size_t)b * T_ * N_OUT + o;
    float*       op = out  + (size_t)b * N_OUT * T_;
    float mem = 0.f;
    for (int tc = 0; tc < T_; tc += 32) {
#pragma unroll
        for (int t = 0; t < 32; ++t) {
            float cur   = cp[(size_t)(tc + t) * N_OUT];
            float reset = (mem - thr > 0.f) ? thr : 0.f;
            mem = beta * mem + cur - reset;
            buf[o * 32 + t] = mem;
        }
        __syncthreads();
#pragma unroll
        for (int it = 0; it < 32; ++it) {
            int e  = it * 256 + threadIdx.x;
            int oo = e >> 5;
            int tt = e & 31;
            op[(size_t)oo * T_ + tc + tt] = buf[oo * 32 + tt];
        }
        __syncthreads();
    }
}

// ---------------------------------------------------------------------------
// Host launcher. Workspace layout:
//   [0,             256MB)  CUR1  f32  [M1, N_HID]
//   [256MB,         384MB)  SPK2  bf16 [M1, N_HID]
//   [384MB,         416MB)  CUR2  f32  [M1, N_OUT]
// ---------------------------------------------------------------------------
extern "C" void kernel_launch(void* const* d_in, const int* in_sizes, int n_in,
                              void* d_out, int out_size, void* d_ws, size_t ws_size,
                              hipStream_t stream)
{
    (void)in_sizes; (void)n_in; (void)out_size; (void)ws_size;
    const float* x     = (const float*)d_in[0];
    const float* W1    = (const float*)d_in[1];
    const float* b1    = (const float*)d_in[2];
    const float* beta2 = (const float*)d_in[3];
    const float* thr2  = (const float*)d_in[4];
    const float* W2    = (const float*)d_in[5];
    const float* b2    = (const float*)d_in[6];
    const float* beta3 = (const float*)d_in[7];
    const float* thr3  = (const float*)d_in[8];
    float* out = (float*)d_out;

    char* ws = (char*)d_ws;
    float*    cur1 = (float*)ws;
    uint16_t* spk2 = (uint16_t*)(ws + (size_t)M1 * N_HID * sizeof(float));
    float*    cur2 = (float*)(ws + (size_t)M1 * N_HID * (sizeof(float) + sizeof(uint16_t)));

    snn_gemm1<<<dim3(M1 / 128, N_HID / 128), 256, 0, stream>>>(x, W1, b1, cur1);
    snn_scan2<<<dim3((B_ * N_HID) / 256), 256, 0, stream>>>(cur1, spk2, beta2, thr2);
    snn_gemm2<<<dim3(M1 / 128, N_OUT / 128), 256, 0, stream>>>(spk2, W2, b2, cur2);
    snn_scan3<<<dim3(B_), 256, 0, stream>>>(cur2, out, beta3, thr3);
}
